// NormalizedEightPointNet_79980880986617
// MI455X (gfx1250) — compile-verified, matrix operand order checked by reading, not run on hardware
//
#include <hip/hip_runtime.h>
#include <hip/hip_bf16.h>

typedef __attribute__((ext_vector_type(16))) _Float16 v16h;
typedef __attribute__((ext_vector_type(8)))  _Float16 v8h;
typedef __attribute__((ext_vector_type(8)))  float    v8f;

#define NPTS   4096
#define NBATCH 8
#define EPS    1e-5f
#define SLOPE  0.01f

// ---------------- block reductions ----------------
__device__ __forceinline__ float bred_sum(float* red, int tid, float v) {
    red[tid] = v; __syncthreads();
    for (int s = 128; s > 0; s >>= 1) {
        if (tid < s) red[tid] += red[tid + s];
        __syncthreads();
    }
    float r = red[0]; __syncthreads();
    return r;
}
__device__ __forceinline__ float bred_max(float* red, int tid, float v) {
    red[tid] = v; __syncthreads();
    for (int s = 128; s > 0; s >>= 1) {
        if (tid < s) red[tid] = fmaxf(red[tid], red[tid + s]);
        __syncthreads();
    }
    float r = red[0]; __syncthreads();
    return r;
}

// ---------------- prep: _rescale + net input ----------------
// one block per batch, 256 threads
__global__ void prep_kernel(const float* __restrict__ pts,   // (B,N,4)
                            float* __restrict__ T1out,       // (B,9)
                            float* __restrict__ T2out,       // (B,9)
                            float* __restrict__ p1,          // (B,N,2) normalized x,y (z==1)
                            float* __restrict__ p2,          // (B,N,2)
                            _Float16* __restrict__ act0)     // (B,N,32) zero-padded
{
    __shared__ float red[256];
    const int b = blockIdx.x, tid = threadIdx.x;
    const float* P = pts + (size_t)b * NPTS * 4;

    float sx1 = 0.f, sy1 = 0.f, sx2 = 0.f, sy2 = 0.f;
    for (int n = tid; n < NPTS; n += 256) {
        sx1 += P[n * 4 + 0]; sy1 += P[n * 4 + 1];
        sx2 += P[n * 4 + 2]; sy2 += P[n * 4 + 3];
    }
    sx1 = bred_sum(red, tid, sx1); sy1 = bred_sum(red, tid, sy1);
    sx2 = bred_sum(red, tid, sx2); sy2 = bred_sum(red, tid, sy2);
    const float c1x = sx1 / NPTS, c1y = sy1 / NPTS;
    const float c2x = sx2 / NPTS, c2y = sy2 / NPTS;

    float sd1 = 0.f, sd2 = 0.f;
    for (int n = tid; n < NPTS; n += 256) {
        float dx1 = P[n * 4 + 0] - c1x, dy1 = P[n * 4 + 1] - c1y;
        float dx2 = P[n * 4 + 2] - c2x, dy2 = P[n * 4 + 3] - c2y;
        sd1 += sqrtf(dx1 * dx1 + dy1 * dy1);
        sd2 += sqrtf(dx2 * dx2 + dy2 * dy2);
    }
    sd1 = bred_sum(red, tid, sd1); sd2 = bred_sum(red, tid, sd2);
    const float s1 = (float)NPTS / sd1;   // 1/meandist
    const float s2 = (float)NPTS / sd2;

    if (tid == 0) {
        float* T1 = T1out + b * 9;
        T1[0] = s1; T1[1] = 0.f; T1[2] = -c1x * s1;
        T1[3] = 0.f; T1[4] = s1; T1[5] = -c1y * s1;
        T1[6] = 0.f; T1[7] = 0.f; T1[8] = 1.f;
        float* T2 = T2out + b * 9;
        T2[0] = s2; T2[1] = 0.f; T2[2] = -c2x * s2;
        T2[3] = 0.f; T2[4] = s2; T2[5] = -c2y * s2;
        T2[6] = 0.f; T2[7] = 0.f; T2[8] = 1.f;
    }

    for (int n = tid; n < NPTS; n += 256) {
        float X1 = s1 * (P[n * 4 + 0] - c1x), Y1 = s1 * (P[n * 4 + 1] - c1y);
        float X2 = s2 * (P[n * 4 + 2] - c2x), Y2 = s2 * (P[n * 4 + 3] - c2y);
        size_t r = (size_t)b * NPTS + n;
        p1[r * 2 + 0] = X1; p1[r * 2 + 1] = Y1;
        p2[r * 2 + 0] = X2; p2[r * 2 + 1] = Y2;
        _Float16* a = act0 + r * 32;
        a[0] = (_Float16)((X1 + 1.f) * 0.5f);
        a[1] = (_Float16)((Y1 + 1.f) * 0.5f);
        a[2] = (_Float16)((X2 + 1.f) * 0.5f);
        a[3] = (_Float16)((Y2 + 1.f) * 0.5f);
        #pragma unroll
        for (int c = 4; c < 32; ++c) a[c] = (_Float16)0.f;
    }
}

// ---------------- weight fp32 -> fp16 (K padded) ----------------
__global__ void wconv_kernel(const float* __restrict__ w, _Float16* __restrict__ w16,
                             int Cout, int Cin, int Kp)
{
    int i = blockIdx.x * 256 + threadIdx.x;
    if (i >= Cout * Kp) return;
    int o = i / Kp, k = i - o * Kp;
    w16[i] = (k < Cin) ? (_Float16)w[o * Cin + k] : (_Float16)0.f;
}

// ---------------- WMMA GEMM: raw[b,n,m] = sum_k W[m,k]*act[b,n,k] + bias[m] ----------------
// grid: (N/512, M/16, B); block 256 = 8 waves; each wave computes a 16(M) x 64(N) tile
// using one shared A-fragment and 4 independent accumulators per K-step.
__global__ void __launch_bounds__(256)
gemm_wmma_kernel(const _Float16* __restrict__ act,  // (B,N,K)
                 const _Float16* __restrict__ w16,  // (M,K)
                 const float* __restrict__ bias,    // (M)
                 float* __restrict__ raw,           // (B,N,M)
                 int M, int K)
{
    const int lane = threadIdx.x & 31;
    const int wave = threadIdx.x >> 5;
    const int b    = blockIdx.z;
    const int m0   = blockIdx.y * 16;
    const int n0   = blockIdx.x * 512 + wave * 64;
    const int half = lane >> 4;         // 0 or 1
    const int kg   = half * 8;          // K sub-offset within fragment
    const int row  = lane & 15;         // M index for A-frag / N index for B-frags

    const _Float16* wrow = w16 + (size_t)(m0 + row) * K;
    const _Float16* arow0 = act + ((size_t)b * NPTS + (n0 +  0 + row)) * (size_t)K;
    const _Float16* arow1 = act + ((size_t)b * NPTS + (n0 + 16 + row)) * (size_t)K;
    const _Float16* arow2 = act + ((size_t)b * NPTS + (n0 + 32 + row)) * (size_t)K;
    const _Float16* arow3 = act + ((size_t)b * NPTS + (n0 + 48 + row)) * (size_t)K;

    v8f c0 = {}, c1 = {}, c2 = {}, c3 = {};
    #pragma unroll 2
    for (int kk = 0; kk < K; kk += 32) {
        // hint the next K-chunk into cache (lowers to global_prefetch_b8)
        __builtin_prefetch(arow0 + kk + 64, 0, 1);
        __builtin_prefetch(arow2 + kk + 64, 0, 1);

        v8h alo = *(const v8h*)(wrow + kk + kg);
        v8h ahi = *(const v8h*)(wrow + kk + 16 + kg);
        v16h A;
        #pragma unroll
        for (int i = 0; i < 8; ++i) { A[i] = alo[i]; A[i + 8] = ahi[i]; }

        v8h b0l = *(const v8h*)(arow0 + kk + kg);
        v8h b0h = *(const v8h*)(arow0 + kk + 16 + kg);
        v8h b1l = *(const v8h*)(arow1 + kk + kg);
        v8h b1h = *(const v8h*)(arow1 + kk + 16 + kg);
        v8h b2l = *(const v8h*)(arow2 + kk + kg);
        v8h b2h = *(const v8h*)(arow2 + kk + 16 + kg);
        v8h b3l = *(const v8h*)(arow3 + kk + kg);
        v8h b3h = *(const v8h*)(arow3 + kk + 16 + kg);
        v16h B0, B1, B2, B3;
        #pragma unroll
        for (int i = 0; i < 8; ++i) {
            B0[i] = b0l[i]; B0[i + 8] = b0h[i];
            B1[i] = b1l[i]; B1[i + 8] = b1h[i];
            B2[i] = b2l[i]; B2[i + 8] = b2h[i];
            B3[i] = b3l[i]; B3[i + 8] = b3h[i];
        }
        c0 = __builtin_amdgcn_wmma_f32_16x16x32_f16(false, A, false, B0, (short)0, c0, false, false);
        c1 = __builtin_amdgcn_wmma_f32_16x16x32_f16(false, A, false, B1, (short)0, c1, false, false);
        c2 = __builtin_amdgcn_wmma_f32_16x16x32_f16(false, A, false, B2, (short)0, c2, false, false);
        c3 = __builtin_amdgcn_wmma_f32_16x16x32_f16(false, A, false, B3, (short)0, c3, false, false);
    }

    // C/D layout: VGPR j -> M = m0 + 8*half + j ; N = ntile + (lane&15)
    const int mo = m0 + half * 8;
    float bv[8];
    #pragma unroll
    for (int j = 0; j < 8; ++j) bv[j] = bias[mo + j];

    v8f* accs[4] = { &c0, &c1, &c2, &c3 };
    #pragma unroll
    for (int t = 0; t < 4; ++t) {
        v8f c = *accs[t];
        float* dst = raw + ((size_t)b * NPTS + (n0 + t * 16 + row)) * (size_t)M + mo;
        float4 r0, r1;
        r0.x = c[0] + bv[0]; r0.y = c[1] + bv[1]; r0.z = c[2] + bv[2]; r0.w = c[3] + bv[3];
        r1.x = c[4] + bv[4]; r1.y = c[5] + bv[5]; r1.z = c[6] + bv[6]; r1.w = c[7] + bv[7];
        *(float4*)(dst)     = r0;
        *(float4*)(dst + 4) = r1;
    }
}

// ---------------- instance norm (over N) + leaky relu, fp32 -> fp16 ----------------
// grid: (C/32, B); block 256 = 8 groups x 32 channels (coalesced along C)
__global__ void instnorm_kernel(const float* __restrict__ raw,   // (B,N,C)
                                const float* __restrict__ g,
                                const float* __restrict__ be,
                                _Float16* __restrict__ act,      // (B,N,C)
                                int C)
{
    __shared__ float s1s[8][32];
    __shared__ float s2s[8][32];
    const int b  = blockIdx.y;
    const int c0 = blockIdx.x * 32;
    const int cs = threadIdx.x & 31;
    const int gq = threadIdx.x >> 5;

    const float* src = raw + (size_t)b * NPTS * C + c0 + cs;
    float s = 0.f, s2 = 0.f;
    for (int n = gq; n < NPTS; n += 8) {
        float x = src[(size_t)n * C];
        s += x; s2 += x * x;
    }
    s1s[gq][cs] = s; s2s[gq][cs] = s2;
    __syncthreads();
    if (gq == 0) {
        float a = 0.f, q = 0.f;
        #pragma unroll
        for (int i = 0; i < 8; ++i) { a += s1s[i][cs]; q += s2s[i][cs]; }
        s1s[0][cs] = a; s2s[0][cs] = q;
    }
    __syncthreads();
    const float mu  = s1s[0][cs] / NPTS;
    const float var = s2s[0][cs] / NPTS - mu * mu;
    const float rs  = rsqrtf(var + EPS);
    const float gg = g[c0 + cs], bb = be[c0 + cs];

    _Float16* dst = act + (size_t)b * NPTS * C + c0 + cs;
    for (int n = gq; n < NPTS; n += 8) {
        float x = src[(size_t)n * C];
        float y = (x - mu) * rs * gg + bb;
        y = (y >= 0.f) ? y : SLOPE * y;
        dst[(size_t)n * C] = (_Float16)y;
    }
}

// ---------------- last layer: 256 -> 1 ----------------
__global__ void lastlayer_kernel(const _Float16* __restrict__ act,  // (B,N,256)
                                 const float* __restrict__ w,       // (1,256)
                                 const float* __restrict__ bias,    // (1)
                                 float* __restrict__ logits)        // (B,N)
{
    const int n = blockIdx.x * 256 + threadIdx.x;
    const int b = blockIdx.y;
    const _Float16* a = act + ((size_t)b * NPTS + n) * 256;
    float s = bias[0];
    #pragma unroll 8
    for (int k = 0; k < 256; ++k) s += (float)a[k] * w[k];
    logits[(size_t)b * NPTS + n] = s;
}

// ---------------- softmax over N ----------------
__global__ void softmax_kernel(const float* __restrict__ logits, float* __restrict__ wout)
{
    __shared__ float red[256];
    const int b = blockIdx.x, tid = threadIdx.x;
    const float* L = logits + (size_t)b * NPTS;
    float mx = -3.4e38f;
    for (int n = tid; n < NPTS; n += 256) mx = fmaxf(mx, L[n]);
    mx = bred_max(red, tid, mx);
    float se = 0.f;
    for (int n = tid; n < NPTS; n += 256) se += __expf(L[n] - mx);
    se = bred_sum(red, tid, se);
    const float inv = 1.f / se;
    for (int n = tid; n < NPTS; n += 256)
        wout[(size_t)b * NPTS + n] = __expf(L[n] - mx) * inv;
}

// ---------------- 9x9 / 3x3 symmetric Jacobi eigensolver ----------------
__device__ void jacobi_sym(float* A, float* V, int n, int sweeps)
{
    for (int i = 0; i < n * n; ++i) V[i] = 0.f;
    for (int i = 0; i < n; ++i) V[i * n + i] = 1.f;
    for (int sw = 0; sw < sweeps; ++sw) {
        for (int p = 0; p < n - 1; ++p) {
            for (int q = p + 1; q < n; ++q) {
                float apq = A[p * n + q];
                if (fabsf(apq) < 1e-30f) continue;
                float app = A[p * n + p], aqq = A[q * n + q];
                float tau = (aqq - app) / (2.f * apq);
                float t = ((tau >= 0.f) ? 1.f : -1.f) / (fabsf(tau) + sqrtf(1.f + tau * tau));
                float c = 1.f / sqrtf(1.f + t * t), s = t * c;
                for (int k = 0; k < n; ++k) {
                    float akp = A[k * n + p], akq = A[k * n + q];
                    A[k * n + p] = c * akp - s * akq;
                    A[k * n + q] = s * akp + c * akq;
                }
                for (int k = 0; k < n; ++k) {
                    float apk = A[p * n + k], aqk = A[q * n + k];
                    A[p * n + k] = c * apk - s * aqk;
                    A[q * n + k] = s * apk + c * aqk;
                }
                for (int k = 0; k < n; ++k) {
                    float vkp = V[k * n + p], vkq = V[k * n + q];
                    V[k * n + p] = c * vkp - s * vkq;
                    V[k * n + q] = s * vkp + c * vkq;
                }
            }
        }
    }
}

// ---------------- eight-point tail: one block per batch ----------------
__global__ void tail_kernel(const float* __restrict__ p1,   // (B,N,2)
                            const float* __restrict__ p2,   // (B,N,2)
                            const float* __restrict__ wsm,  // (B,N)
                            float* __restrict__ Fout)       // (B,9)
{
    __shared__ float red[256];
    const int b = blockIdx.x, tid = threadIdx.x;
    const float* w  = wsm + (size_t)b * NPTS;
    const float* P1 = p1 + (size_t)b * NPTS * 2;
    const float* P2 = p2 + (size_t)b * NPTS * 2;

    float sw = 0.f, sx1 = 0.f, sy1 = 0.f, sx2 = 0.f, sy2 = 0.f;
    for (int n = tid; n < NPTS; n += 256) {
        float wn = w[n];
        sw  += wn;
        sx1 += wn * P1[n * 2 + 0]; sy1 += wn * P1[n * 2 + 1];
        sx2 += wn * P2[n * 2 + 0]; sy2 += wn * P2[n * 2 + 1];
    }
    sw  = bred_sum(red, tid, sw);
    sx1 = bred_sum(red, tid, sx1); sy1 = bred_sum(red, tid, sy1);
    sx2 = bred_sum(red, tid, sx2); sy2 = bred_sum(red, tid, sy2);
    const float c1x = sx1 / sw, c1y = sy1 / sw;
    const float c2x = sx2 / sw, c2y = sy2 / sw;

    float sd1 = 0.f, sd2 = 0.f;
    for (int n = tid; n < NPTS; n += 256) {
        float wn = w[n];
        float dx1 = P1[n * 2 + 0] - c1x, dy1 = P1[n * 2 + 1] - c1y;
        float dx2 = P2[n * 2 + 0] - c2x, dy2 = P2[n * 2 + 1] - c2y;
        sd1 += wn * sqrtf(dx1 * dx1 + dy1 * dy1);
        sd2 += wn * sqrtf(dx2 * dx2 + dy2 * dy2);
    }
    sd1 = bred_sum(red, tid, sd1); sd2 = bred_sum(red, tid, sd2);
    const float s1 = 1.4142f * sw / sd1;    // 1.4142 / meandist
    const float s2 = 1.4142f * sw / sd2;

    // weighted Gram of the 9-vector design rows: G = sum_n w^2 v v^T
    float G[45];
    #pragma unroll
    for (int i = 0; i < 45; ++i) G[i] = 0.f;
    for (int n = tid; n < NPTS; n += 256) {
        float a1 = s1 * (P1[n * 2 + 0] - c1x), b1 = s1 * (P1[n * 2 + 1] - c1y);
        float a2 = s2 * (P2[n * 2 + 0] - c2x), b2 = s2 * (P2[n * 2 + 1] - c2y);
        float v[9] = { a1 * a2, a1 * b2, a1,
                       b1 * a2, b1 * b2, b1,
                       a2,      b2,      1.f };
        float w2 = w[n] * w[n];
        int idx = 0;
        #pragma unroll
        for (int i = 0; i < 9; ++i)
            #pragma unroll
            for (int j = i; j < 9; ++j)
                G[idx++] += w2 * v[i] * v[j];
    }
    for (int k = 0; k < 45; ++k) G[k] = bred_sum(red, tid, G[k]);

    if (tid == 0) {
        float A[81], V[81];
        int idx = 0;
        for (int i = 0; i < 9; ++i)
            for (int j = i; j < 9; ++j) {
                A[i * 9 + j] = G[idx]; A[j * 9 + i] = G[idx]; ++idx;
            }
        jacobi_sym(A, V, 9, 15);
        int im = 0; float mv = A[0];
        for (int i = 1; i < 9; ++i) {
            float d = A[i * 9 + i];
            if (d < mv) { mv = d; im = i; }
        }
        float F[9];
        for (int i = 0; i < 9; ++i) F[i] = V[i * 9 + im];

        // rank-2 projection via smallest right singular vector of F
        float M3[9], V3[9];
        for (int i = 0; i < 3; ++i)
            for (int j = 0; j < 3; ++j) {
                float acc = 0.f;
                for (int k = 0; k < 3; ++k) acc += F[k * 3 + i] * F[k * 3 + j];
                M3[i * 3 + j] = acc;
            }
        jacobi_sym(M3, V3, 3, 12);
        int i3 = 0; float m3 = M3[0];
        for (int i = 1; i < 3; ++i) {
            float d = M3[i * 3 + i];
            if (d < m3) { m3 = d; i3 = i; }
        }
        float v3[3] = { V3[0 * 3 + i3], V3[1 * 3 + i3], V3[2 * 3 + i3] };
        float Fv[3];
        for (int i = 0; i < 3; ++i)
            Fv[i] = F[i * 3 + 0] * v3[0] + F[i * 3 + 1] * v3[1] + F[i * 3 + 2] * v3[2];
        float Fp[9];
        for (int i = 0; i < 3; ++i)
            for (int j = 0; j < 3; ++j)
                Fp[i * 3 + j] = F[i * 3 + j] - Fv[i] * v3[j];

        float t1[9] = { s1, 0.f, -c1x * s1, 0.f, s1, -c1y * s1, 0.f, 0.f, 1.f };
        float t2[9] = { s2, 0.f, -c2x * s2, 0.f, s2, -c2y * s2, 0.f, 0.f, 1.f };
        // out = t1^T * Fp * t2
        float tmp[9];
        for (int i = 0; i < 3; ++i)
            for (int k = 0; k < 3; ++k) {
                float acc = 0.f;
                for (int j = 0; j < 3; ++j) acc += t1[j * 3 + i] * Fp[j * 3 + k];
                tmp[i * 3 + k] = acc;
            }
        for (int i = 0; i < 3; ++i)
            for (int l = 0; l < 3; ++l) {
                float acc = 0.f;
                for (int k = 0; k < 3; ++k) acc += tmp[i * 3 + k] * t2[k * 3 + l];
                Fout[b * 9 + i * 3 + l] = acc;
            }
    }
}

// ---------------- host launcher ----------------
extern "C" void kernel_launch(void* const* d_in, const int* in_sizes, int n_in,
                              void* d_out, int out_size, void* d_ws, size_t ws_size,
                              hipStream_t stream) {
    (void)in_sizes; (void)n_in; (void)out_size; (void)ws_size;
    const float* pts = (const float*)d_in[0];
    const float* cw[6] = { (const float*)d_in[2],  (const float*)d_in[6],
                           (const float*)d_in[10], (const float*)d_in[14],
                           (const float*)d_in[18], (const float*)d_in[22] };
    const float* cb[6] = { (const float*)d_in[3],  (const float*)d_in[7],
                           (const float*)d_in[11], (const float*)d_in[15],
                           (const float*)d_in[19], (const float*)d_in[23] };
    const float* g[5]  = { (const float*)d_in[4],  (const float*)d_in[8],
                           (const float*)d_in[12], (const float*)d_in[16],
                           (const float*)d_in[20] };
    const float* be[5] = { (const float*)d_in[5],  (const float*)d_in[9],
                           (const float*)d_in[13], (const float*)d_in[17],
                           (const float*)d_in[21] };
    float* out = (float*)d_out;   // [Fmat 72 | T1 72 | T2 72 | weights 32768]

    char* ws = (char*)d_ws;
    size_t o = 0;
    float*    raw  = (float*)(ws + o);    o += (size_t)NBATCH * NPTS * 1024 * 4;  // 128 MiB
    _Float16* actA = (_Float16*)(ws + o); o += (size_t)NBATCH * NPTS * 1024 * 2;  //  64 MiB
    _Float16* actB = (_Float16*)(ws + o); o += (size_t)NBATCH * NPTS * 1024 * 2;  //  64 MiB
    const int Cin[5] = { 4, 64, 128, 1024, 512 };
    const int Kp [5] = { 32, 64, 128, 1024, 512 };
    const int Co [5] = { 64, 128, 1024, 512, 256 };
    _Float16* w16[5];
    for (int i = 0; i < 5; ++i) {
        w16[i] = (_Float16*)(ws + o);
        o += (size_t)Co[i] * Kp[i] * 2;
    }
    float* p1     = (float*)(ws + o); o += (size_t)NBATCH * NPTS * 2 * 4;
    float* p2     = (float*)(ws + o); o += (size_t)NBATCH * NPTS * 2 * 4;
    float* logits = (float*)(ws + o); o += (size_t)NBATCH * NPTS * 4;

    // 1) rescale + T1/T2 + fp16 net input
    prep_kernel<<<NBATCH, 256, 0, stream>>>(pts, out + 72, out + 144, p1, p2, actA);

    // 2) weight conversion (fp32 -> fp16, K padded to >=32)
    for (int i = 0; i < 5; ++i) {
        int tot = Co[i] * Kp[i];
        wconv_kernel<<<(tot + 255) / 256, 256, 0, stream>>>(cw[i], w16[i], Co[i], Cin[i], Kp[i]);
    }

    // 3) five WMMA GEMM + instance-norm layers (ping-pong actA/actB)
    const int NX = NPTS / 512;   // 8 blocks of 512 points (8 waves x 64)
    gemm_wmma_kernel<<<dim3(NX,   64 / 16, NBATCH), 256, 0, stream>>>(actA, w16[0], cb[0], raw,   64,   32);
    instnorm_kernel <<<dim3(  64 / 32, NBATCH), 256, 0, stream>>>(raw, g[0], be[0], actB,   64);
    gemm_wmma_kernel<<<dim3(NX,  128 / 16, NBATCH), 256, 0, stream>>>(actB, w16[1], cb[1], raw,  128,   64);
    instnorm_kernel <<<dim3( 128 / 32, NBATCH), 256, 0, stream>>>(raw, g[1], be[1], actA,  128);
    gemm_wmma_kernel<<<dim3(NX, 1024 / 16, NBATCH), 256, 0, stream>>>(actA, w16[2], cb[2], raw, 1024,  128);
    instnorm_kernel <<<dim3(1024 / 32, NBATCH), 256, 0, stream>>>(raw, g[2], be[2], actB, 1024);
    gemm_wmma_kernel<<<dim3(NX,  512 / 16, NBATCH), 256, 0, stream>>>(actB, w16[3], cb[3], raw,  512, 1024);
    instnorm_kernel <<<dim3( 512 / 32, NBATCH), 256, 0, stream>>>(raw, g[3], be[3], actA,  512);
    gemm_wmma_kernel<<<dim3(NX,  256 / 16, NBATCH), 256, 0, stream>>>(actA, w16[4], cb[4], raw,  256,  512);
    instnorm_kernel <<<dim3( 256 / 32, NBATCH), 256, 0, stream>>>(raw, g[4], be[4], actB,  256);

    // 4) final 256->1 layer, softmax -> weights output
    lastlayer_kernel<<<dim3(NPTS / 256, NBATCH), 256, 0, stream>>>(actB, cw[5], cb[5], logits);
    softmax_kernel<<<NBATCH, 256, 0, stream>>>(logits, out + 216);

    // 5) weighted eight-point tail -> Fmat
    tail_kernel<<<NBATCH, 256, 0, stream>>>(p1, p2, out + 216, out);
}
